// SpatialChannelContext_74887049773168
// MI455X (gfx1250) — compile-verified, matrix operand order checked by reading, not run on hardware
//
#include <hip/hip_runtime.h>
#include <hip/hip_bf16.h>

#define B_  4
#define C_  256
#define H_  256
#define W_  256
#define P_  64
#define N_  4096   // 64*64
#define C8_ 32
#define R_  16
#define VROW_ 40   // padded LDS row (halfs): 80B -> conflict-free b128 reads

typedef __attribute__((ext_vector_type(16))) _Float16 v16h;
typedef __attribute__((ext_vector_type(8)))  _Float16 v8h;
typedef __attribute__((ext_vector_type(8)))  float    v8f;

// ---------------------------------------------------------------------------
// D = A(16x32) x B(32x16) + C, f16 in / f32 accum
// ---------------------------------------------------------------------------
__device__ __forceinline__ v8f wmma_f16(v16h a, v16h b, v8f c) {
  return __builtin_amdgcn_wmma_f32_16x16x32_f16(
      /*neg_a=*/false, a, /*neg_b=*/false, b,
      /*c_mod=*/(short)0, c, /*reuse_a=*/false, /*reuse_b=*/false);
}

// Load one lane's 16 K-elements from a row of 32 contiguous halfs.
// A layout (16x32, 16-bit): lane half h: e0..7 -> k = h*8+e ; e8..15 -> k = 16+h*8+(e-8)
// B layout (32x16) mirrors it per-column. base must be 16B aligned.
__device__ __forceinline__ v16h load_row32(const _Float16* __restrict__ base, int half) {
  v8h lo = *(const v8h*)(base + (half << 3));
  v8h hi = *(const v8h*)(base + 16 + (half << 3));
  v16h r;
#pragma unroll
  for (int e = 0; e < 8; ++e) { r[e] = lo[e]; r[8 + e] = hi[e]; }
  return r;
}

// async global -> LDS copy of one 64B V row (4 x b128), ASYNCcnt-tracked.
// LDS byte address = low 32 bits of the generic pointer (wave-relative).
__device__ __forceinline__ void async_copy_row64B(const _Float16* __restrict__ src,
                                                  _Float16* dst) {
  unsigned       l = (unsigned)(unsigned long long)(uintptr_t)dst;
  unsigned long long g = (unsigned long long)(uintptr_t)src;
#pragma unroll
  for (int j = 0; j < 4; ++j) {
    asm volatile("global_load_async_to_lds_b128 %0, %1, off"
                 :: "v"(l + j * 16), "v"(g + j * 16) : "memory");
  }
}

// ---------------------------------------------------------------------------
// 1) AdaptiveAvgPool2d(64): xin [B,C,256,256] -> xf [B,C,4096] (4x4 block mean)
// ---------------------------------------------------------------------------
__global__ void k_pool(const float* __restrict__ xin, float* __restrict__ xf) {
  int idx = blockIdx.x * blockDim.x + threadIdx.x;       // B*C*64*64
  if (idx >= B_ * C_ * P_ * P_) return;
  int px = idx & 63, py = (idx >> 6) & 63, bc = idx >> 12;
  const float* base = xin + ((size_t)bc * H_ + py * 4) * W_ + px * 4;
  float s = 0.f;
#pragma unroll
  for (int r = 0; r < 4; ++r) {
    float4 v = *(const float4*)(base + (size_t)r * W_);
    s += v.x + v.y + v.z + v.w;
  }
  xf[(size_t)bc * N_ + (py << 6) + px] = s * 0.0625f;
}

// ---------------------------------------------------------------------------
// 2) transpose + f16: xf [B,C,N] f32 -> xh [B,N,C] f16  (32x32 LDS tiles)
// ---------------------------------------------------------------------------
__global__ void k_transpose_cn_nc(const float* __restrict__ xf, _Float16* __restrict__ xh) {
  __shared__ float tile[32][33];
  int blk = blockIdx.x;
  int nt = blk & 127, ct = (blk >> 7) & 7, b = blk >> 10;
  int tx = threadIdx.x & 31, ty0 = threadIdx.x >> 5;
#pragma unroll
  for (int i = 0; i < 4; ++i) {
    int ty = ty0 + i * 8;                                 // c within tile
    tile[ty][tx] = xf[((size_t)(b * C_ + ct * 32 + ty)) * N_ + nt * 32 + tx];
  }
  __syncthreads();
#pragma unroll
  for (int i = 0; i < 4; ++i) {
    int ty = ty0 + i * 8;                                 // n within tile
    xh[((size_t)(b * N_ + nt * 32 + ty)) * C_ + ct * 32 + tx] = (_Float16)tile[tx][ty];
  }
}

__global__ void k_f32_to_f16(const float* __restrict__ s, _Float16* __restrict__ d, int n) {
  int i = blockIdx.x * blockDim.x + threadIdx.x;
  if (i < n) d[i] = (_Float16)s[i];
}

// ---------------------------------------------------------------------------
// 3) QKV GEMM: Out[n][o] = sum_c X[b][n][c] * W[o][c] + bias[o]
//    transposed==0 : store f16 [B][N][OC]   (q, k^T)
//    transposed==1 : store f16 [B][OC][N]   (v, via per-wave LDS transpose)
//    one 16x16 output tile per wave, K=256 -> 8 WMMAs
// ---------------------------------------------------------------------------
__global__ void k_gemm_xw(const _Float16* __restrict__ X, const _Float16* __restrict__ W,
                          const float* __restrict__ bias, _Float16* __restrict__ out,
                          int OC, int transposed) {
  __shared__ _Float16 lds[8][16][17];
  int wid = threadIdx.x >> 5, lane = threadIdx.x & 31;
  int half = lane >> 4, l16 = lane & 15;
  int gw = blockIdx.x * 8 + wid;
  int otiles = OC >> 4;
  int ot = gw % otiles, ntb = gw / otiles;
  int nt = ntb & (N_ / 16 - 1), b = ntb / (N_ / 16);
  int n0 = nt * 16, o0 = ot * 16;

  const _Float16* xrow = X + ((size_t)(b * N_ + n0 + l16)) * C_;   // A: row n
  const _Float16* wrow = W + (size_t)(o0 + l16) * C_;              // B: col o
  v8f acc = {};
#pragma unroll
  for (int c0 = 0; c0 < C_; c0 += 32) {
    v16h a = load_row32(xrow + c0, half);
    v16h w = load_row32(wrow + c0, half);
    acc = wmma_f16(a, w, acc);
  }
  float bs = bias[o0 + l16];
  if (!transposed) {
#pragma unroll
    for (int r = 0; r < 8; ++r) {
      int nl = half * 8 + r;
      out[((size_t)(b * N_ + n0 + nl)) * OC + o0 + l16] = (_Float16)(acc[r] + bs);
    }
  } else {
#pragma unroll
    for (int r = 0; r < 8; ++r) lds[wid][half * 8 + r][l16] = (_Float16)(acc[r] + bs);
    __syncthreads();
#pragma unroll
    for (int r = 0; r < 8; ++r) {
      int ol = half * 8 + r;
      out[((size_t)(b * OC + o0 + ol)) * N_ + n0 + l16] = lds[wid][l16][ol];
    }
  }
}

// ---------------------------------------------------------------------------
// 4) Fused flash attention + residual-1.
//    E^T = K' x q^T  (accumulator layout == B layout of next WMMA)
//    O^T = V x P^T ; s1[b][c][m] = xf + gamma1 * O^T[c][m]/l
//    One wave per 16-row m-tile. V chunk (16KB) is shared by all 8 waves of a
//    block -> double-buffered async global->LDS staging (ASYNCcnt pipeline),
//    8x less L2 traffic than per-wave streaming.
// ---------------------------------------------------------------------------
__global__ void __launch_bounds__(256)
k_attn_flash(const _Float16* __restrict__ qh,   // [B][N][32]
             const _Float16* __restrict__ kth,  // [B][N][32]
             const _Float16* __restrict__ vh,   // [B][C][N]
             const float* __restrict__ xf,      // [B][C][N]
             const float* __restrict__ g1p,
             float* __restrict__ s1) {
  __shared__ _Float16 vtile[2][C_][VROW_];      // 2 x 256 rows x 80B = 40KB

  int tid = threadIdx.x;
  int gw = blockIdx.x * (blockDim.x >> 5) + (tid >> 5);
  int lane = tid & 31, half = lane >> 4, l16 = lane & 15;
  int mt = gw & (N_ / 16 - 1), b = gw / (N_ / 16);   // b uniform per block
  int m0 = mt * 16;
  float gamma1 = g1p[0];

  // this thread stages V row c = tid (all waves share batch b)
  const _Float16* vrow = vh + ((size_t)(b * C_ + tid)) * N_;

  // q^T B-operand: lane column m = m0+l16, invariant across the key loop
  v16h qB = load_row32(qh + ((size_t)(b * N_ + m0 + l16)) * C8_, half);

  v8f acc[16];
  v8f vz = {};
#pragma unroll
  for (int t = 0; t < 16; ++t) acc[t] = vz;
  float run_max = -1e30f, run_sum = 0.f;

  // prologue: stage chunk 0
  async_copy_row64B(vrow, &vtile[0][tid][0]);

  constexpr int NCH = N_ / 32;
  for (int ic = 0; ic < NCH; ++ic) {
    int buf = ic & 1;
    int n0 = ic * 32;
    bool more = (ic + 1 < NCH);

    if (more) {
      async_copy_row64B(vrow + n0 + 32, &vtile[buf ^ 1][tid][0]);  // prefetch next
      asm volatile("s_wait_asynccnt 0x4" ::: "memory");            // current chunk done
    } else {
      asm volatile("s_wait_asynccnt 0x0" ::: "memory");
    }
    __syncthreads();                                               // all rows visible

    if (n0 + 64 < N_)
      __builtin_prefetch(kth + ((size_t)(b * N_ + n0 + 64 + l16)) * C8_, 0, 1);

    // E^T tiles (n-rows x m-cols), K = 32 = C8 in one WMMA each
    v16h kA0 = load_row32(kth + ((size_t)(b * N_ + n0 + l16)) * C8_, half);
    v16h kA1 = load_row32(kth + ((size_t)(b * N_ + n0 + 16 + l16)) * C8_, half);
    v8f e0 = wmma_f16(kA0, qB, vz);
    v8f e1 = wmma_f16(kA1, qB, vz);

    // online softmax: lane owns 16 of the 32 n's for its m; partner is lane^16
    float mx = -1e30f;
#pragma unroll
    for (int r = 0; r < 8; ++r) mx = fmaxf(mx, fmaxf(e0[r], e1[r]));
    mx = fmaxf(mx, __shfl_xor(mx, 16, 32));
    float nm = fmaxf(run_max, mx);
    float corr = __expf(run_max - nm);
    float lsum = 0.f;
    v16h p;                                // already in B-operand layout for PV
#pragma unroll
    for (int r = 0; r < 8; ++r) {
      float p0 = __expf(e0[r] - nm);
      float p1 = __expf(e1[r] - nm);
      lsum += p0 + p1;
      p[r] = (_Float16)p0;
      p[8 + r] = (_Float16)p1;
    }
    lsum += __shfl_xor(lsum, 16, 32);
    run_sum = run_sum * corr + lsum;
    run_max = nm;
#pragma unroll
    for (int t = 0; t < 16; ++t)
#pragma unroll
      for (int r = 0; r < 8; ++r) acc[t][r] *= corr;

    // O^T += V(16c x 32n) x P^T(32n x 16m), V from LDS (conflict-free rows)
#pragma unroll
    for (int t = 0; t < 16; ++t) {
      const _Float16* row = &vtile[buf][t * 16 + l16][0];
      v8h lo = *(const v8h*)(row + (half << 3));
      v8h hi = *(const v8h*)(row + 16 + (half << 3));
      v16h vA;
#pragma unroll
      for (int e = 0; e < 8; ++e) { vA[e] = lo[e]; vA[8 + e] = hi[e]; }
      acc[t] = wmma_f16(vA, p, acc[t]);
    }
    __syncthreads();   // everyone done with vtile[buf] before it is re-staged
  }

  float inv = 1.f / run_sum;
  int m = m0 + l16;
#pragma unroll
  for (int t = 0; t < 16; ++t)
#pragma unroll
    for (int r = 0; r < 8; ++r) {
      int c = t * 16 + half * 8 + r;
      size_t idx = ((size_t)(b * C_ + c)) * N_ + m;      // 16 lanes -> 64B segment
      s1[idx] = xf[idx] + gamma1 * (acc[t][r] * inv);
    }
}

// ---------------------------------------------------------------------------
// 5) SE path
// ---------------------------------------------------------------------------
__global__ void k_rowmean(const float* __restrict__ s1, float* __restrict__ chn) {
  __shared__ float red[256];
  int row = blockIdx.x;                                   // b*C + c
  const float* p = s1 + (size_t)row * N_;
  float s = 0.f;
  for (int i = threadIdx.x; i < N_; i += 256) s += p[i];
  red[threadIdx.x] = s;
  __syncthreads();
  for (int st = 128; st > 0; st >>= 1) {
    if (threadIdx.x < st) red[threadIdx.x] += red[threadIdx.x + st];
    __syncthreads();
  }
  if (threadIdx.x == 0) chn[row] = red[0] * (1.f / N_);
}

__global__ void k_se(const float* __restrict__ chn,
                     const float* __restrict__ W1, const float* __restrict__ b1,
                     const float* __restrict__ W2, const float* __restrict__ b2,
                     float* __restrict__ se) {
  __shared__ float cs[256];
  __shared__ float hid[16];
  int b = blockIdx.x, t = threadIdx.x;
  cs[t] = chn[b * C_ + t];
  __syncthreads();
  if (t < R_) {
    float s = b1[t];
    for (int c = 0; c < C_; ++c) s += W1[t * C_ + c] * cs[c];
    hid[t] = fmaxf(s, 0.f);
  }
  __syncthreads();
  float s = b2[t];
#pragma unroll
  for (int r = 0; r < R_; ++r) s += W2[t * R_ + r] * hid[r];
  se[b * C_ + t] = 1.f / (1.f + __expf(-s));
}

// ---------------------------------------------------------------------------
// 6) residual-2:  feature = xf + gamma2 * (s1 * se[b][c])
// ---------------------------------------------------------------------------
__global__ void k_feature(const float* __restrict__ xf, const float* __restrict__ s1,
                          const float* __restrict__ se, const float* __restrict__ g2p,
                          float* __restrict__ feat) {
  int idx = blockIdx.x * blockDim.x + threadIdx.x;
  if (idx >= B_ * C_ * N_) return;
  int bc = idx >> 12;                                     // /N_
  feat[idx] = xf[idx] + g2p[0] * (s1[idx] * se[bc]);
}

// ---------------------------------------------------------------------------
// 7) bilinear 4x upsample 64->256, half-pixel centers, edge clamp
// ---------------------------------------------------------------------------
__global__ void k_upsample(const float* __restrict__ feat, float* __restrict__ out) {
  long long idx = (long long)blockIdx.x * 256 + threadIdx.x;
  if (idx >= (long long)B_ * C_ * H_ * W_) return;
  int j = (int)(idx & 255), i = (int)((idx >> 8) & 255);
  int plane = (int)(idx >> 16);
  float sx = (j + 0.5f) * 0.25f - 0.5f;
  float sy = (i + 0.5f) * 0.25f - 0.5f;
  float fxw = sx - floorf(sx), fyw = sy - floorf(sy);
  int x0 = (int)floorf(sx), y0 = (int)floorf(sy);
  int x1 = min(x0 + 1, P_ - 1), y1 = min(y0 + 1, P_ - 1);
  x0 = max(x0, 0); y0 = max(y0, 0);
  const float* pl = feat + (size_t)plane * N_;
  float v00 = pl[y0 * P_ + x0], v01 = pl[y0 * P_ + x1];
  float v10 = pl[y1 * P_ + x0], v11 = pl[y1 * P_ + x1];
  out[idx] = (1.f - fyw) * ((1.f - fxw) * v00 + fxw * v01) +
             fyw * ((1.f - fxw) * v10 + fxw * v11);
}

// ---------------------------------------------------------------------------
extern "C" void kernel_launch(void* const* d_in, const int* in_sizes, int n_in,
                              void* d_out, int out_size, void* d_ws, size_t ws_size,
                              hipStream_t stream) {
  const float* xin = (const float*)d_in[0];
  const float* Wq  = (const float*)d_in[1];
  const float* bq  = (const float*)d_in[2];
  const float* Wk  = (const float*)d_in[3];
  const float* bk  = (const float*)d_in[4];
  const float* Wv  = (const float*)d_in[5];
  const float* bv  = (const float*)d_in[6];
  const float* W1  = (const float*)d_in[7];
  const float* b1  = (const float*)d_in[8];
  const float* W2  = (const float*)d_in[9];
  const float* b2  = (const float*)d_in[10];
  const float* g1  = (const float*)d_in[11];
  const float* g2  = (const float*)d_in[12];
  float* out = (float*)d_out;

  char* wsp = (char*)d_ws;
  auto take = [&](size_t bytes) -> char* {
    char* p = wsp;
    wsp += (bytes + 255) & ~(size_t)255;
    return p;
  };
  float*     xf    = (float*)take((size_t)B_ * C_ * N_ * 4);       // pooled
  float*     s1    = (float*)take((size_t)B_ * C_ * N_ * 4);       // residual-1
  float*     feat  = (float*)take((size_t)B_ * C_ * N_ * 4);       // residual-2
  _Float16*  xh    = (_Float16*)take((size_t)B_ * N_ * C_ * 2);    // [B][N][C]
  _Float16*  vhb   = (_Float16*)take((size_t)B_ * C_ * N_ * 2);    // [B][C][N]
  _Float16*  qh    = (_Float16*)take((size_t)B_ * N_ * C8_ * 2);   // [B][N][32]
  _Float16*  kth   = (_Float16*)take((size_t)B_ * N_ * C8_ * 2);   // [B][N][32]
  _Float16*  Wq16  = (_Float16*)take((size_t)C8_ * C_ * 2);
  _Float16*  Wk16  = (_Float16*)take((size_t)C8_ * C_ * 2);
  _Float16*  Wv16  = (_Float16*)take((size_t)C_ * C_ * 2);
  float*     chn   = (float*)take((size_t)B_ * C_ * 4);
  float*     seb   = (float*)take((size_t)B_ * C_ * 4);

  // weights -> f16
  k_f32_to_f16<<<(C8_ * C_ + 255) / 256, 256, 0, stream>>>(Wq, Wq16, C8_ * C_);
  k_f32_to_f16<<<(C8_ * C_ + 255) / 256, 256, 0, stream>>>(Wk, Wk16, C8_ * C_);
  k_f32_to_f16<<<(C_ * C_ + 255) / 256, 256, 0, stream>>>(Wv, Wv16, C_ * C_);

  // pool + layout
  k_pool<<<(B_ * C_ * P_ * P_) / 256, 256, 0, stream>>>(xin, xf);
  k_transpose_cn_nc<<<B_ * (C_ / 32) * (N_ / 32), 256, 0, stream>>>(xf, xh);

  // q, k^T : [B][N][32] ; v : [B][C][N]
  k_gemm_xw<<<(B_ * (N_ / 16) * (C8_ / 16)) / 8, 256, 0, stream>>>(xh, Wq16, bq, qh,  C8_, 0);
  k_gemm_xw<<<(B_ * (N_ / 16) * (C8_ / 16)) / 8, 256, 0, stream>>>(xh, Wk16, bk, kth, C8_, 0);
  k_gemm_xw<<<(B_ * (N_ / 16) * (C_  / 16)) / 8, 256, 0, stream>>>(xh, Wv16, bv, vhb, C_,  1);

  // fused attention + residual-1 (async-LDS double-buffered V)
  k_attn_flash<<<(B_ * (N_ / 16)) / 8, 256, 0, stream>>>(qh, kth, vhb, xf, g1, s1);

  // SE gate + residual-2
  k_rowmean<<<B_ * C_, 256, 0, stream>>>(s1, chn);
  k_se<<<B_, 256, 0, stream>>>(chn, W1, b1, W2, b2, seb);
  k_feature<<<(B_ * C_ * N_) / 256, 256, 0, stream>>>(xf, s1, seb, g2, feat);

  // upsample to [B,C,256,256]
  k_upsample<<<(B_ * C_ * H_ * W_) / 256, 256, 0, stream>>>(feat, out);
}
